// GritTransformerLayer_84954453115474
// MI455X (gfx1250) — matrix operand order repaired and checked by reading, not compiled, not run on hardware
//
#include <hip/hip_runtime.h>
#include <hip/hip_bf16.h>

typedef __attribute__((ext_vector_type(16))) _Float16 v16h;
typedef __attribute__((ext_vector_type(8)))  _Float16 v8h;
typedef __attribute__((ext_vector_type(4)))  _Float16 v4h;
typedef __attribute__((ext_vector_type(8)))  float    v8f;
typedef __attribute__((ext_vector_type(4)))  float    v4f;

#define GRIT_N 40000
#define GRIT_E 640000
#define GRIT_D 128
#define GRIT_H 8
#define GRIT_DH 16

// ---------------------------------------------------------------------------
// Utility kernels
// ---------------------------------------------------------------------------
__global__ void fill_zero_kernel(float* __restrict__ p, size_t n) {
    size_t i = (size_t)blockIdx.x * blockDim.x + threadIdx.x;
    size_t stride = (size_t)gridDim.x * blockDim.x;
    for (; i < n; i += stride) p[i] = 0.f;
}

// n must be a multiple of 4 (all our tensors are multiples of 128)
__global__ void f32_to_f16_kernel(const float* __restrict__ in, _Float16* __restrict__ out, size_t n4) {
    size_t i = (size_t)blockIdx.x * blockDim.x + threadIdx.x;
    size_t stride = (size_t)gridDim.x * blockDim.x;
    const v4f* in4 = (const v4f*)in;
    v4h* out4 = (v4h*)out;
    for (; i < n4; i += stride) {
        v4f v = in4[i];
        v4h o;
        o[0] = (_Float16)v[0]; o[1] = (_Float16)v[1];
        o[2] = (_Float16)v[2]; o[3] = (_Float16)v[3];
        out4[i] = o;
    }
}

// Pre-swizzle a row-major f32 weight W[K x Nc] into per-lane WMMA B fragments.
// Bp[(ntile*KT + kt)*512 + lane*16 + j] = W[kt*32 + (lane>=16?16:0) + j][ntile*16 + (lane&15)]
__global__ void pack_b_kernel(const float* __restrict__ W, _Float16* __restrict__ Bp, int K, int Nc) {
    int idx = blockIdx.x * blockDim.x + threadIdx.x;
    int total = K * Nc;
    if (idx >= total) return;
    int j    = idx & 15;
    int lane = (idx >> 4) & 31;
    int tile = idx >> 9;            // ntile*KT + kt
    int KT   = K >> 5;
    int kt   = tile % KT;
    int ntile= tile / KT;
    int n = ntile * 16 + (lane & 15);
    int k = kt * 32 + ((lane >> 4) * 16) + j;
    Bp[idx] = (_Float16)W[(size_t)k * Nc + n];
}

// ---------------------------------------------------------------------------
// WMMA GEMM: C[M x Nc] = A_f16[M x K] @ B_packed (+bias) (+resid) (relu?)
// Block = 8 waves, each wave owns a 16x64 strip (4 n-tiles). The shared B
// strip (4*KT*512 halves, <= 32 KB) is staged once into LDS; the hot loop is
// A-only global loads (2 x b128 per k-tile) + ds_load B fragments feeding
// 4 back-to-back WMMAs. Requires Nc % 64 == 0, M % 16 == 0.
// ---------------------------------------------------------------------------
__global__ void wmma_gemm_kernel(const _Float16* __restrict__ A,
                                 const _Float16* __restrict__ Bp,
                                 const float* __restrict__ bias,
                                 const float* __restrict__ resid,
                                 float* __restrict__ Cf,
                                 _Float16* __restrict__ Ch,
                                 int M, int K, int Nc, int relu) {
    __shared__ _Float16 Bs[16384];               // up to KT=8: 4*8*512 halves = 32 KB
    const int KT     = K >> 5;
    const int ntile0 = blockIdx.y * 4;

    // ---- cooperative stage of the block's B strip (contiguous in Bp) ----
    {
        const v8h* src = (const v8h*)(Bp + (size_t)ntile0 * KT * 512);
        v8h* dst = (v8h*)Bs;
        const int chunks = KT * 256;             // (4*KT*512)/8
        for (int i = threadIdx.x; i < chunks; i += blockDim.x)
            dst[i] = src[i];
    }
    __syncthreads();                             // only barrier in kernel

    const int lane  = threadIdx.x & 31;
    const int wv    = threadIdx.x >> 5;
    const int mtile = blockIdx.x * 8 + wv;
    if (mtile * 16 >= M) return;                 // uniform per wave, after barrier

    const int m    = mtile * 16 + (lane & 15);
    const int koff = (lane >> 4) * 16;           // A layout: hi lanes take +16 halves

    v8f acc0 = {}, acc1 = {}, acc2 = {}, acc3 = {};
    const _Float16* aptr  = A  + (size_t)m * K + koff;
    const _Float16* bbase = Bs + lane * 16;
    const int bstride = KT * 512;                // halves between n-tiles in LDS

    for (int kt = 0; kt < KT; ++kt) {
        union { v16h v; v8h h[2]; } a;
        a.h[0] = *(const v8h*)(aptr);            // K = kt*32 + koff .. +7
        a.h[1] = *(const v8h*)(aptr + 8);        // K = kt*32 + 8 + koff .. +7
        const _Float16* bp = bbase + kt * 512;
        v16h b0 = *(const v16h*)(bp);
        v16h b1 = *(const v16h*)(bp + bstride);
        v16h b2 = *(const v16h*)(bp + 2 * bstride);
        v16h b3 = *(const v16h*)(bp + 3 * bstride);
        acc0 = __builtin_amdgcn_wmma_f32_16x16x32_f16(false, a.v, false, b0, (short)0, acc0, false, false);
        acc1 = __builtin_amdgcn_wmma_f32_16x16x32_f16(false, a.v, false, b1, (short)0, acc1, false, false);
        acc2 = __builtin_amdgcn_wmma_f32_16x16x32_f16(false, a.v, false, b2, (short)0, acc2, false, false);
        acc3 = __builtin_amdgcn_wmma_f32_16x16x32_f16(false, a.v, false, b3, (short)0, acc3, false, false);
        aptr += 32;
    }

    const int rowbase = mtile * 16 + ((lane >> 4) << 3);   // lanes 16-31 hold M=r+8
    v8f accs[4] = {acc0, acc1, acc2, acc3};
    #pragma unroll
    for (int nt = 0; nt < 4; ++nt) {
        const int n = (ntile0 + nt) * 16 + (lane & 15);
        const float bb = bias ? bias[n] : 0.f;
        #pragma unroll
        for (int r = 0; r < 8; ++r) {
            int row = rowbase + r;
            size_t off = (size_t)row * Nc + n;
            float v = accs[nt][r] + bb;
            if (resid) v += resid[off];
            if (relu)  v = fmaxf(v, 0.f);
            if (Cf) Cf[off] = v;
            if (Ch) Ch[off] = (_Float16)v;
        }
    }
}

// ---------------------------------------------------------------------------
// Edge pass 1: score, signed-sqrt, relu -> e_t (stored f16), logit, den atomics
// One thread per (edge, head); vectorized v4f / v8h accesses.
// ---------------------------------------------------------------------------
__global__ void edge_pass1_kernel(const int* __restrict__ ei,
                                  const float* __restrict__ Kh,
                                  const float* __restrict__ Qh,
                                  const _Float16* __restrict__ Eeh,
                                  const float* __restrict__ Aw,
                                  _Float16* __restrict__ wEh,
                                  float* __restrict__ logitB,
                                  float* __restrict__ den) {
    int idx = blockIdx.x * blockDim.x + threadIdx.x;
    if (idx >= GRIT_E * GRIT_H) return;
    int h = idx & 7;
    int e = idx >> 3;
    int src = ei[e];
    int dst = ei[GRIT_E + e];

    const v4f* kp = (const v4f*)(Kh + (size_t)src * GRIT_D + h * GRIT_DH);
    const v4f* qp = (const v4f*)(Qh + (size_t)dst * GRIT_D + h * GRIT_DH);
    const _Float16* ew = Eeh + (size_t)e * (2 * GRIT_D) + h * (2 * GRIT_DH); // [16 w | 16 b]

    float kq[GRIT_DH];
    #pragma unroll
    for (int t = 0; t < 4; ++t) {
        v4f a = kp[t], b = qp[t];
        kq[4 * t + 0] = a[0] + b[0];
        kq[4 * t + 1] = a[1] + b[1];
        kq[4 * t + 2] = a[2] + b[2];
        kq[4 * t + 3] = a[3] + b[3];
    }
    union { v8h v; _Float16 s[8]; } w0, w1, b0, b1, o0, o1;
    w0.v = *(const v8h*)(ew);
    w1.v = *(const v8h*)(ew + 8);
    b0.v = *(const v8h*)(ew + 16);
    b1.v = *(const v8h*)(ew + 24);

    float logit = 0.f;
    #pragma unroll
    for (int j = 0; j < GRIT_DH; ++j) {
        float wj = (j < 8) ? (float)w0.s[j & 7] : (float)w1.s[j & 7];
        float bj = (j < 8) ? (float)b0.s[j & 7] : (float)b1.s[j & 7];
        float s  = kq[j] * wj;
        float ss = (s > 0.f) ? sqrtf(s) : ((s < 0.f) ? -sqrtf(-s) : 0.f);
        float et = fmaxf(ss + bj, 0.f);
        if (j < 8) o0.s[j & 7] = (_Float16)et; else o1.s[j & 7] = (_Float16)et;
        logit += et * Aw[j * GRIT_H + h];
    }
    _Float16* wp = wEh + (size_t)e * GRIT_D + h * GRIT_DH;
    *(v8h*)(wp)     = o0.v;
    *(v8h*)(wp + 8) = o1.v;

    logit = fminf(fmaxf(logit, -5.f), 5.f);
    logitB[(size_t)e * GRIT_H + h] = logit;
    // CLAMP=5 => exp bounded in [6.7e-3, 149]; segment-max folds away safely in f32.
    atomicAdd(&den[(size_t)dst * GRIT_H + h], __expf(logit));
}

// ---------------------------------------------------------------------------
// Edge pass 2: attn-weighted scatter of Vh[src] and e_t into wV / rowV.
// Accumulators (41 MB) live in the 192 MB L2, so atomics stay on-chip.
// ---------------------------------------------------------------------------
__global__ void edge_pass2_kernel(const int* __restrict__ ei,
                                  const float* __restrict__ Vh,
                                  const _Float16* __restrict__ wEh,
                                  const float* __restrict__ logitB,
                                  const float* __restrict__ den,
                                  float* __restrict__ wV,
                                  float* __restrict__ rowV) {
    int idx = blockIdx.x * blockDim.x + threadIdx.x;
    if (idx >= GRIT_E * GRIT_H) return;
    int h = idx & 7;
    int e = idx >> 3;
    int src = ei[e];
    int dst = ei[GRIT_E + e];

    float attn = __expf(logitB[(size_t)e * GRIT_H + h]) /
                 (den[(size_t)dst * GRIT_H + h] + 1e-16f);

    const v4f* vp = (const v4f*)(Vh + (size_t)src * GRIT_D + h * GRIT_DH);
    const _Float16* wp = wEh + (size_t)e * GRIT_D + h * GRIT_DH;
    union { v8h v; _Float16 s[8]; } w0, w1;
    w0.v = *(const v8h*)(wp);
    w1.v = *(const v8h*)(wp + 8);

    float vv[GRIT_DH];
    #pragma unroll
    for (int t = 0; t < 4; ++t) {
        v4f a = vp[t];
        vv[4 * t + 0] = a[0]; vv[4 * t + 1] = a[1];
        vv[4 * t + 2] = a[2]; vv[4 * t + 3] = a[3];
    }
    float* wVp = wV   + (size_t)dst * GRIT_D + h * GRIT_DH;
    float* rVp = rowV + (size_t)dst * GRIT_D + h * GRIT_DH;
    #pragma unroll
    for (int j = 0; j < GRIT_DH; ++j) {
        float et = (j < 8) ? (float)w0.s[j & 7] : (float)w1.s[j & 7];
        atomicAdd(&wVp[j], vv[j] * attn);
        atomicAdd(&rVp[j], et * attn);
    }
}

// ---------------------------------------------------------------------------
// Node: wV + einsum(rowV, VeRow) then degree scaling -> f16 for WOh GEMM.
// One thread per (node, head).
// ---------------------------------------------------------------------------
__global__ void node_kernel(const float* __restrict__ wV,
                            const float* __restrict__ rowV,
                            const float* __restrict__ VeRow,
                            const float* __restrict__ deg_coef,
                            const float* __restrict__ log_deg,
                            _Float16* __restrict__ hpreh) {
    int idx = blockIdx.x * blockDim.x + threadIdx.x;
    if (idx >= GRIT_N * GRIT_H) return;
    int h = idx & 7;
    int n = idx >> 3;
    float rv[GRIT_DH];
    #pragma unroll
    for (int d = 0; d < GRIT_DH; ++d)
        rv[d] = rowV[(size_t)n * GRIT_D + h * GRIT_DH + d];
    float ld = log_deg[n];
    union { v8h v; _Float16 s[8]; } o0, o1;
    #pragma unroll
    for (int c = 0; c < GRIT_DH; ++c) {
        float acc = wV[(size_t)n * GRIT_D + h * GRIT_DH + c];
        #pragma unroll
        for (int d = 0; d < GRIT_DH; ++d)
            acc += rv[d] * VeRow[d * GRIT_D + h * GRIT_DH + c];   // VeRow[d][h][c]
        int col = h * GRIT_DH + c;
        float hv = acc * (deg_coef[col * 2] + ld * deg_coef[col * 2 + 1]);
        if (c < 8) o0.s[c & 7] = (_Float16)hv; else o1.s[c & 7] = (_Float16)hv;
    }
    _Float16* hp = hpreh + (size_t)n * GRIT_D + h * GRIT_DH;
    *(v8h*)(hp)     = o0.v;
    *(v8h*)(hp + 8) = o1.v;
}

// ---------------------------------------------------------------------------
// BatchNorm over rows (axis 0): coalesced row-slab partial sums + atomics.
// blockDim.x == 128 (one thread per column).
// ---------------------------------------------------------------------------
__global__ void col_partial_kernel(const float* __restrict__ buf,
                                   float* __restrict__ sums,  // [128 sum | 128 sumsq]
                                   int M, int rowsPerBlock) {
    int c = threadIdx.x;
    size_t r0 = (size_t)blockIdx.x * rowsPerBlock;
    size_t r1 = r0 + rowsPerBlock;
    if (r1 > (size_t)M) r1 = M;
    float s = 0.f, ss = 0.f;
    for (size_t r = r0; r < r1; ++r) {
        float v = buf[r * GRIT_D + c];
        s += v; ss += v * v;
    }
    atomicAdd(&sums[c], s);
    atomicAdd(&sums[GRIT_D + c], ss);
}

__global__ void bn_finalize_kernel(const float* __restrict__ sums,
                                   float* __restrict__ stats, float invM) {
    int c = threadIdx.x;
    float mean = sums[c] * invM;
    float var  = sums[GRIT_D + c] * invM - mean * mean;
    stats[c] = mean;
    stats[GRIT_D + c] = rsqrtf(var + 1e-5f);
}

__global__ void bn_apply_kernel(const float* __restrict__ in,
                                const float* __restrict__ stats,
                                const float* __restrict__ g,
                                const float* __restrict__ b,
                                float* __restrict__ outF,
                                _Float16* __restrict__ outH, size_t total4) {
    size_t i = (size_t)blockIdx.x * blockDim.x + threadIdx.x;
    size_t stride = (size_t)gridDim.x * blockDim.x;
    const v4f* in4 = (const v4f*)in;
    for (; i < total4; i += stride) {
        int c = (int)((i * 4) & (GRIT_D - 1));
        v4f v = in4[i];
        v4f o;
        #pragma unroll
        for (int t = 0; t < 4; ++t)
            o[t] = g[c + t] * (v[t] - stats[c + t]) * stats[GRIT_D + c + t] + b[c + t];
        if (outF) ((v4f*)outF)[i] = o;
        if (outH) {
            v4h oh;
            #pragma unroll
            for (int t = 0; t < 4; ++t) oh[t] = (_Float16)o[t];
            ((v4h*)outH)[i] = oh;
        }
    }
}

// ---------------------------------------------------------------------------
// Host-side orchestration
// ---------------------------------------------------------------------------
extern "C" void kernel_launch(void* const* d_in, const int* in_sizes, int n_in,
                              void* d_out, int out_size, void* d_ws, size_t ws_size,
                              hipStream_t stream) {
    (void)in_sizes; (void)n_in; (void)out_size; (void)ws_size;
    const int N = GRIT_N, E = GRIT_E, D = GRIT_D;

    const float* x         = (const float*)d_in[0];
    const float* edge_attr = (const float*)d_in[1];
    const int*   ei        = (const int*)  d_in[2];
    const float* log_deg   = (const float*)d_in[3];
    const float* Wq  = (const float*)d_in[4];  const float* bq  = (const float*)d_in[5];
    const float* Wk  = (const float*)d_in[6];  const float* bk  = (const float*)d_in[7];
    const float* We  = (const float*)d_in[8];  const float* be  = (const float*)d_in[9];
    const float* Wv  = (const float*)d_in[10]; const float* bv  = (const float*)d_in[11];
    const float* Aw  = (const float*)d_in[12]; const float* VeRow = (const float*)d_in[13];
    const float* WOh = (const float*)d_in[14]; const float* bOh = (const float*)d_in[15];
    const float* WOe = (const float*)d_in[16]; const float* bOe = (const float*)d_in[17];
    const float* deg_coef = (const float*)d_in[18];
    const float* g1h = (const float*)d_in[19]; const float* b1h = (const float*)d_in[20];
    const float* g1e = (const float*)d_in[21]; const float* b1e = (const float*)d_in[22];
    const float* g2h = (const float*)d_in[23]; const float* b2h = (const float*)d_in[24];
    const float* W1  = (const float*)d_in[25]; const float* b1  = (const float*)d_in[26];
    const float* W2  = (const float*)d_in[27]; const float* b2  = (const float*)d_in[28];

    // ---- workspace bump allocator (256 B aligned) ----
    char* base = (char*)d_ws;
    size_t off = 0;
    auto alloc = [&](size_t bytes) -> char* {
        char* p = base + off;
        off += (bytes + 255) & ~(size_t)255;
        return p;
    };
    _Float16* xh    = (_Float16*)alloc((size_t)N * D * 2);
    _Float16* eah   = (_Float16*)alloc((size_t)E * D * 2);
    float*    Qh    = (float*)   alloc((size_t)N * D * 4);
    float*    Kh    = (float*)   alloc((size_t)N * D * 4);
    float*    Vh    = (float*)   alloc((size_t)N * D * 4);
    _Float16* Eeh   = (_Float16*)alloc((size_t)E * 2 * D * 2);
    _Float16* wEh   = (_Float16*)alloc((size_t)E * D * 2);
    float*    logitB= (float*)   alloc((size_t)E * GRIT_H * 4);
    float*    den   = (float*)   alloc((size_t)N * GRIT_H * 4);
    float*    wV    = (float*)   alloc((size_t)N * D * 4);
    float*    rowV  = (float*)   alloc((size_t)N * D * 4);
    _Float16* hpreh = (_Float16*)alloc((size_t)N * D * 2);
    float*    hbuf1 = (float*)   alloc((size_t)N * D * 4);   // x + WOh-out
    float*    hbuf2 = (float*)   alloc((size_t)N * D * 4);   // BN1 f32
    _Float16* h1h   = (_Float16*)alloc((size_t)N * D * 2);   // BN1 f16
    _Float16* acth  = (_Float16*)alloc((size_t)N * 2 * D * 2);
    float*    hbuf3 = (float*)   alloc((size_t)N * D * 4);   // h1 + h2
    float*    sums1 = (float*)   alloc(256 * 4);
    float*    sums2 = (float*)   alloc(256 * 4);
    float*    sums3 = (float*)   alloc(256 * 4);
    float*    stat1 = (float*)   alloc(256 * 4);
    float*    stat2 = (float*)   alloc(256 * 4);
    float*    stat3 = (float*)   alloc(256 * 4);
    _Float16* Wq_p  = (_Float16*)alloc((size_t)D * D * 2);
    _Float16* Wk_p  = (_Float16*)alloc((size_t)D * D * 2);
    _Float16* Wv_p  = (_Float16*)alloc((size_t)D * D * 2);
    _Float16* We_p  = (_Float16*)alloc((size_t)D * 2 * D * 2);
    _Float16* WOh_p = (_Float16*)alloc((size_t)D * D * 2);
    _Float16* WOe_p = (_Float16*)alloc((size_t)D * D * 2);
    _Float16* W1_p  = (_Float16*)alloc((size_t)D * 2 * D * 2);
    _Float16* W2_p  = (_Float16*)alloc((size_t)2 * D * D * 2);

    float* e_out = (float*)d_out + (size_t)N * D;   // e result region
    float* h_out = (float*)d_out;                   // h result region

    // ---- zero atomic accumulators ----
    fill_zero_kernel<<<256, 256, 0, stream>>>(den,  (size_t)N * GRIT_H);
    fill_zero_kernel<<<512, 256, 0, stream>>>(wV,   (size_t)N * D);
    fill_zero_kernel<<<512, 256, 0, stream>>>(rowV, (size_t)N * D);
    fill_zero_kernel<<<1,   256, 0, stream>>>(sums1, 256);
    fill_zero_kernel<<<1,   256, 0, stream>>>(sums2, 256);
    fill_zero_kernel<<<1,   256, 0, stream>>>(sums3, 256);

    // ---- f16 casts of activations ----
    f32_to_f16_kernel<<<2048, 256, 0, stream>>>(x,         xh,  (size_t)N * D / 4);
    f32_to_f16_kernel<<<8192, 256, 0, stream>>>(edge_attr, eah, (size_t)E * D / 4);

    // ---- pack weights into WMMA B-fragment layout ----
    auto pack = [&](const float* W, _Float16* Bp, int K, int Nc) {
        int total = K * Nc;
        pack_b_kernel<<<(total + 255) / 256, 256, 0, stream>>>(W, Bp, K, Nc);
    };
    pack(Wq, Wq_p, D, D);     pack(Wk, Wk_p, D, D);     pack(Wv, Wv_p, D, D);
    pack(We, We_p, D, 2 * D); pack(WOh, WOh_p, D, D);   pack(WOe, WOe_p, D, D);
    pack(W1, W1_p, D, 2 * D); pack(W2, W2_p, 2 * D, D);

    auto gemm = [&](const _Float16* A, const _Float16* Bp, const float* bias,
                    const float* resid, float* Cf, _Float16* Ch,
                    int M, int K, int Nc, int relu) {
        dim3 grid((M / 16 + 7) / 8, Nc / 64);
        wmma_gemm_kernel<<<grid, 256, 0, stream>>>(A, Bp, bias, resid, Cf, Ch, M, K, Nc, relu);
    };

    // ---- Q/K/V projections (N x 128) ----
    gemm(xh, Wq_p, bq, nullptr, Qh, nullptr, N, D, D, 0);
    gemm(xh, Wk_p, bk, nullptr, Kh, nullptr, N, D, D, 0);
    gemm(xh, Wv_p, bv, nullptr, Vh, nullptr, N, D, D, 0);

    // ---- Edge feature projection Ee = edge_attr @ We + be (f16 out) ----
    gemm(eah, We_p, be, nullptr, nullptr, Eeh, E, D, 2 * D, 0);

    // ---- Edge attention passes ----
    int eThreads = E * GRIT_H;
    edge_pass1_kernel<<<(eThreads + 255) / 256, 256, 0, stream>>>(
        ei, Kh, Qh, Eeh, Aw, wEh, logitB, den);
    edge_pass2_kernel<<<(eThreads + 255) / 256, 256, 0, stream>>>(
        ei, Vh, wEh, logitB, den, wV, rowV);

    // ---- Node combine + degree scaler ----
    int nThreads = N * GRIT_H;
    node_kernel<<<(nThreads + 255) / 256, 256, 0, stream>>>(
        wV, rowV, VeRow, deg_coef, log_deg, hpreh);

    // ---- Output projections with residuals ----
    gemm(hpreh, WOh_p, bOh, x,         hbuf1, nullptr, N, D, D, 0);  // x + h@WOh
    gemm(wEh,   WOe_p, bOe, edge_attr, e_out, nullptr, E, D, D, 0);  // ea + wE@WOe

    // ---- BN1h on hbuf1 -> hbuf2 (f32) + h1h (f16) ----
    col_partial_kernel<<<(N + 2047) / 2048, GRIT_D, 0, stream>>>(hbuf1, sums1, N, 2048);
    bn_finalize_kernel<<<1, GRIT_D, 0, stream>>>(sums1, stat1, 1.f / N);
    bn_apply_kernel<<<2048, 256, 0, stream>>>(hbuf1, stat1, g1h, b1h, hbuf2, h1h, (size_t)N * D / 4);

    // ---- BN1e on e_out in place (final e) ----
    col_partial_kernel<<<(E + 2047) / 2048, GRIT_D, 0, stream>>>(e_out, sums2, E, 2048);
    bn_finalize_kernel<<<1, GRIT_D, 0, stream>>>(sums2, stat2, 1.f / E);
    bn_apply_kernel<<<4096, 256, 0, stream>>>(e_out, stat2, g1e, b1e, e_out, nullptr, (size_t)E * D / 4);

    // ---- FFN: act = relu(h1 @ W1 + b1); h2 = act @ W2 + b2 + h1 ----
    gemm(h1h,  W1_p, b1, nullptr, nullptr, acth, N, D,     2 * D, 1);
    gemm(acth, W2_p, b2, hbuf2,   hbuf3,   nullptr, N, 2 * D, D,  0);

    // ---- BN2h -> final h into d_out ----
    col_partial_kernel<<<(N + 2047) / 2048, GRIT_D, 0, stream>>>(hbuf3, sums3, N, 2048);
    bn_finalize_kernel<<<1, GRIT_D, 0, stream>>>(sums3, stat3, 1.f / N);
    bn_apply_kernel<<<2048, 256, 0, stream>>>(hbuf3, stat3, g2h, b2h, h_out, nullptr, (size_t)N * D / 4);
}
